// MiniPointNetOurs_81192061764450
// MI455X (gfx1250) — matrix operand order repaired and checked by reading, not compiled, not run on hardware
//
#include <hip/hip_runtime.h>
#include <hip/hip_bf16.h>

typedef __attribute__((ext_vector_type(16))) _Float16 v16h;
typedef __attribute__((ext_vector_type(8)))  float    v8f;
typedef __attribute__((ext_vector_type(4)))  float    f32x4;
typedef int v4i_ __attribute__((vector_size(16)));

#define BB   32
#define NN   8192
#define KK   64
#define OUTC 384
#define EPSB 1e-5f
#define ENC_NEG_INF 0x007FFFFFu   // enc(-inf)

#if defined(__has_builtin)
#if __has_builtin(__builtin_amdgcn_global_load_async_to_lds_b128)
#define USE_ASYNC_LDS 1
#endif
#endif
#ifndef USE_ASYNC_LDS
#define USE_ASYNC_LDS 0
#endif

// order-preserving float <-> uint32 map (monotonic), lets us use u32 atomic max
__device__ __forceinline__ unsigned encf(float f){
  unsigned u = __float_as_uint(f);
  return (u & 0x80000000u) ? ~u : (u | 0x80000000u);
}
__device__ __forceinline__ float decf(unsigned e){
  unsigned b = (e & 0x80000000u) ? (e ^ 0x80000000u) : ~e;
  return __uint_as_float(b);
}

union FragU { v16h h; f32x4 f[2]; };

// A-fragment (16x32 f16): halves 0..7 at p0, halves 8..15 at p0+16 (same row)
__device__ __forceinline__ v16h ldfragA(const _Float16* p0){
  FragU u;
  u.f[0] = *(const f32x4*)(p0);
  u.f[1] = *(const f32x4*)(p0 + 16);
  return u.h;
}
// B-fragment (32x16 f16): halves 0..7 at p0, halves 8..15 at p0+16 (same weight row)
__device__ __forceinline__ v16h ldfragB(const _Float16* p0){
  FragU u;
  u.f[0] = *(const f32x4*)(p0);
  u.f[1] = *(const f32x4*)(p0 + 16);
  return u.h;
}
__device__ __forceinline__ v8f wmma16(v16h a, v16h b, v8f c){
  return __builtin_amdgcn_wmma_f32_16x16x32_f16(false, a, false, b, (short)0, c, false, false);
}

// 16-byte global -> LDS copy: async DMA (ASYNCcnt) when available, else via VGPRs
__device__ __forceinline__ void cp16(const char* g, char* l){
#if USE_ASYNC_LDS
  __builtin_amdgcn_global_load_async_to_lds_b128(
      (__attribute__((address_space(1))) v4i_*)(g),
      (__attribute__((address_space(3))) v4i_*)(l), 0, 0);
#else
  *(f32x4*)l = *(const f32x4*)g;
#endif
}
__device__ __forceinline__ void async_fence(){
#if USE_ASYNC_LDS
#if __has_builtin(__builtin_amdgcn_s_wait_asynccnt)
  __builtin_amdgcn_s_wait_asynccnt(0);
#else
  asm volatile("s_wait_asynccnt 0x0" ::: "memory");
#endif
#endif
}

// ---------------- prep: fold BN into weights, convert to f16 row-major ----------------
__global__ void k_prep(const float* __restrict__ W1, const float* __restrict__ b1,
                       const float* __restrict__ g1, const float* __restrict__ be1,
                       const float* __restrict__ m1, const float* __restrict__ v1,
                       const float* __restrict__ W2,
                       const float* __restrict__ W3, const float* __restrict__ b3,
                       const float* __restrict__ g2, const float* __restrict__ be2,
                       const float* __restrict__ m2, const float* __restrict__ v2,
                       const float* __restrict__ W4,
                       float* __restrict__ W1f, float* __restrict__ b1f,
                       float* __restrict__ b3f,
                       _Float16* __restrict__ W2h, _Float16* __restrict__ W3h,
                       _Float16* __restrict__ W4h)
{
  int idx = blockIdx.x * 256 + threadIdx.x;   // grid covers 262144
  if (idx < 384){ int d = idx / 3; float s = g1[d] * rsqrtf(v1[d] + EPSB); W1f[idx] = W1[idx] * s; }
  if (idx < 128){ float s = g1[idx] * rsqrtf(v1[idx] + EPSB); b1f[idx] = s * (b1[idx] - m1[idx]) + be1[idx]; }
  if (idx < 512){ float s = g2[idx] * rsqrtf(v2[idx] + EPSB); b3f[idx] = s * (b3[idx] - m2[idx]) + be2[idx]; }
  if (idx < 256*128) W2h[idx] = (_Float16)W2[idx];
  if (idx < 512*512){ int o = idx >> 9; float s = g2[o] * rsqrtf(v2[o] + EPSB); W3h[idx] = (_Float16)(W3[idx] * s); }
  if (idx < 384*512) W4h[idx] = (_Float16)W4[idx];
}

// ---------------- init encoded segment-max buffers ----------------
__global__ void k_init(unsigned* __restrict__ fg_enc, unsigned* __restrict__ out_enc)
{
  int idx = blockIdx.x * 256 + threadIdx.x;   // grid covers 786432
  if (idx < BB*KK*256)  fg_enc[idx]  = ENC_NEG_INF;
  if (idx < BB*KK*OUTC) out_enc[idx] = ENC_NEG_INF;
}

// ---------------- stage 1: h=ReLU(BN(xyz@W1)) (VALU), feat=h@W2+b2 (WMMA) ----------------
// block = 256 threads (8 waves), 128 points per workgroup, 1 M-tile of 16 pts per wave
// W2h (64KB) is staged LDS-resident once per workgroup via async copy.
__global__ __launch_bounds__(256) void k_stage1(
    const float* __restrict__ xyz, const int* __restrict__ choice,
    const float* __restrict__ W1f, const float* __restrict__ b1f,
    const _Float16* __restrict__ W2h, const float* __restrict__ b2,
    _Float16* __restrict__ feat_h, unsigned* __restrict__ fg_enc)
{
  __shared__ _Float16 hT[128][128];    // 32 KB: per-point 128-ch hidden, f16
  __shared__ _Float16 w2s[256*128];    // 64 KB: full W2 in LDS
  __shared__ float w1s[384];
  __shared__ float b1s[128];
  int tid = threadIdx.x;

  // kick off the async W2 copy first so the DMA overlaps the VALU conv below
  {
    const char* src = (const char*)W2h;
    char* dst = (char*)w2s;
    #pragma unroll
    for (int i = 0; i < 16; ++i)
      cp16(src + tid*16 + i*4096, dst + tid*16 + i*4096);
  }

  for (int i = tid; i < 384; i += 256) w1s[i] = W1f[i];
  if (tid < 128) b1s[tid] = b1f[tid];
  __syncthreads();

  long base = (long)blockIdx.x * 128;          // 8192 % 128 == 0 -> tile within one batch
  { // tiny 3->128 conv + folded BN + ReLU, 2 threads per point (64 channels each)
    int p = tid & 127, half = tid >> 7;
    long gp = base + p;
    float x = xyz[gp*3+0], y = xyz[gp*3+1], z = xyz[gp*3+2];
    for (int d = half*64; d < half*64 + 64; ++d){
      float v = fmaf(x, w1s[d*3+0], fmaf(y, w1s[d*3+1], fmaf(z, w1s[d*3+2], b1s[d])));
      hT[p][d] = (_Float16)fmaxf(v, 0.f);
    }
  }
  async_fence();
  __syncthreads();

  int lane = tid & 31, wave = tid >> 5;
  int Ml = lane & 15, hi = lane >> 4;
  int mbase = wave * 16;
  int bidx = (int)(base / NN);

  v16h a[4];                                   // K = 128 -> 4 chunks
  #pragma unroll
  for (int c = 0; c < 4; ++c)
    a[c] = ldfragA(&hT[mbase + Ml][c*32 + hi*8]);

  int segr[8]; long gptr[8];
  #pragma unroll
  for (int r = 0; r < 8; ++r){
    gptr[r] = base + mbase + r + 8*hi;         // C-frag row: M = r + 8*hi
    segr[r] = bidx * KK + choice[gptr[r]];
  }

  for (int nt = 0; nt < 16; ++nt){             // 256 output channels
    v8f acc = {};
    #pragma unroll
    for (int c = 0; c < 4; ++c)
      acc = wmma16(a[c], ldfragB(&w2s[(16*nt + Ml)*128 + c*32 + hi*8]), acc);
    float bias = b2[16*nt + Ml];
    #pragma unroll
    for (int r = 0; r < 8; ++r){
      float v = acc[r] + bias;
      feat_h[gptr[r]*256 + 16*nt + Ml] = (_Float16)v;
      atomicMax(&fg_enc[(long)segr[r]*256 + 16*nt + Ml], encf(v));
    }
  }
}

// ---------------- decode cluster maxima to f16 rows for stage-2 gather ----------------
__global__ void k_decfg(const unsigned* __restrict__ fg_enc, _Float16* __restrict__ fgh)
{
  int idx = blockIdx.x * 256 + threadIdx.x;    // grid covers 524288
  if (idx < BB*KK*256) fgh[idx] = (_Float16)decf(fg_enc[idx]);
}

// ---------------- stage 2: x=[fg_pts|feat] (512) -> ReLU(BN(x@W3)) -> @W4 -> seg max ----
// block = 128 threads (4 waves), each wave one 16-point M-tile, 64 points / WG.
// Weight tiles (16 rows x 512 halves = 16KB, contiguous) are double-buffered in LDS via
// async global->LDS copies so one copy serves all 4 waves and overlaps the WMMAs.
__global__ __launch_bounds__(128) void k_stage2(
    const int* __restrict__ choice,
    const _Float16* __restrict__ feat_h, const _Float16* __restrict__ fgh,
    const _Float16* __restrict__ W3h, const float* __restrict__ b3f,
    const _Float16* __restrict__ W4h, const float* __restrict__ b4,
    unsigned* __restrict__ out_enc)
{
  __shared__ _Float16 mid[4][16][512];         // 64 KB: ReLU'd 512-ch intermediate per wave
  __shared__ _Float16 wb[2][16*512];           // 32 KB: double-buffered weight tile
  int tid = threadIdx.x;
  int lane = tid & 31, wave = tid >> 5;
  int Ml = lane & 15, hi = lane >> 4;
  long base = (long)blockIdx.x * 64 + wave * 16;
  int bidx = (int)(base / NN);

  const char* w3c = (const char*)W3h;
  const char* w4c = (const char*)W4h;
  char* wb0 = (char*)&wb[0][0];
  char* wb1 = (char*)&wb[1][0];

  // kick off W3 tile 0 copy (16KB, flat)
  #pragma unroll
  for (int i = 0; i < 8; ++i)
    cp16(w3c + tid*16 + i*2048, wb0 + tid*16 + i*2048);

  long ptA = base + Ml;                        // A-frag row for this lane
  int segA = bidx * KK + choice[ptA];

  v16h a[16];                                  // K = 512 resident in VGPRs
  const _Float16* fgrow = fgh + (long)segA * 256;      // channels 0..255: gathered cluster max
  #pragma unroll
  for (int c = 0; c < 8; ++c) a[c] = ldfragA(fgrow + c*32 + hi*8);
  const _Float16* ftrow = feat_h + ptA * 256;          // channels 256..511: per-point feat
  #pragma unroll
  for (int c = 0; c < 8; ++c) a[8 + c] = ldfragA(ftrow + c*32 + hi*8);

  int segr[8];
  #pragma unroll
  for (int r = 0; r < 8; ++r) segr[r] = bidx * KK + choice[base + r + 8*hi];

  async_fence();
  __syncthreads();

  for (int nt = 0; nt < 32; ++nt){             // 512 outputs of matmul1
    char* cur = (nt & 1) ? wb1 : wb0;
    char* nxt = (nt & 1) ? wb0 : wb1;
    if (nt + 1 < 32){                          // prefetch next weight tile into other buffer
      #pragma unroll
      for (int i = 0; i < 8; ++i)
        cp16(w3c + (size_t)(nt + 1)*16384 + tid*16 + i*2048, nxt + tid*16 + i*2048);
    }
    const _Float16* wt = (const _Float16*)cur;
    v8f acc = {};
    #pragma unroll
    for (int c = 0; c < 16; ++c)
      acc = wmma16(a[c], ldfragB(wt + Ml*512 + c*32 + hi*8), acc);
    float bias = b3f[16*nt + Ml];
    #pragma unroll
    for (int r = 0; r < 8; ++r)
      mid[wave][r + 8*hi][16*nt + Ml] = (_Float16)fmaxf(acc[r] + bias, 0.f);
    async_fence();
    __syncthreads();
  }

  // kick off W4 tile 0 copy; reload A-frags from LDS mid (ds_load_b128)
  #pragma unroll
  for (int i = 0; i < 8; ++i)
    cp16(w4c + tid*16 + i*2048, wb0 + tid*16 + i*2048);
  #pragma unroll
  for (int c = 0; c < 16; ++c)
    a[c] = ldfragA(&mid[wave][Ml][c*32 + hi*8]);
  async_fence();
  __syncthreads();

  for (int nt = 0; nt < 24; ++nt){             // 384 outputs of matmul2
    char* cur = (nt & 1) ? wb1 : wb0;
    char* nxt = (nt & 1) ? wb0 : wb1;
    if (nt + 1 < 24){
      #pragma unroll
      for (int i = 0; i < 8; ++i)
        cp16(w4c + (size_t)(nt + 1)*16384 + tid*16 + i*2048, nxt + tid*16 + i*2048);
    }
    const _Float16* wt = (const _Float16*)cur;
    v8f acc = {};
    #pragma unroll
    for (int c = 0; c < 16; ++c)
      acc = wmma16(a[c], ldfragB(wt + Ml*512 + c*32 + hi*8), acc);
    float bias = b4[16*nt + Ml];
    #pragma unroll
    for (int r = 0; r < 8; ++r)
      atomicMax(&out_enc[(long)segr[r]*OUTC + 16*nt + Ml], encf(acc[r] + bias));
    async_fence();
    __syncthreads();
  }
}

// ---------------- decode final segmented max to fp32 output ----------------
__global__ void k_decout(const unsigned* __restrict__ out_enc, float* __restrict__ out)
{
  int idx = blockIdx.x * 256 + threadIdx.x;    // grid covers 786432
  if (idx < BB*KK*OUTC) out[idx] = decf(out_enc[idx]);
}

extern "C" void kernel_launch(void* const* d_in, const int* in_sizes, int n_in,
                              void* d_out, int out_size, void* d_ws, size_t ws_size,
                              hipStream_t stream)
{
  const float* xyz    = (const float*)d_in[0];
  const int*   choice = (const int*)  d_in[1];
  const float* W1  = (const float*)d_in[2];
  const float* b1  = (const float*)d_in[3];
  const float* g1  = (const float*)d_in[4];
  const float* be1 = (const float*)d_in[5];
  const float* m1  = (const float*)d_in[6];
  const float* v1  = (const float*)d_in[7];
  const float* W2  = (const float*)d_in[8];
  const float* b2  = (const float*)d_in[9];
  const float* W3  = (const float*)d_in[10];
  const float* b3  = (const float*)d_in[11];
  const float* g2  = (const float*)d_in[12];
  const float* be2 = (const float*)d_in[13];
  const float* m2  = (const float*)d_in[14];
  const float* v2  = (const float*)d_in[15];
  const float* W4  = (const float*)d_in[16];
  const float* b4  = (const float*)d_in[17];

  char* ws = (char*)d_ws;
  size_t o = 0;
  auto take = [&](size_t n) -> char* {
    char* p = ws + o;
    o += (n + 255) & ~(size_t)255;
    return p;
  };
  float*    W1f     = (float*)   take(384 * 4);
  float*    b1f     = (float*)   take(128 * 4);
  float*    b3f     = (float*)   take(512 * 4);
  _Float16* W2h     = (_Float16*)take((size_t)256*128*2);
  _Float16* W3h     = (_Float16*)take((size_t)512*512*2);
  _Float16* W4h     = (_Float16*)take((size_t)384*512*2);
  unsigned* fg_enc  = (unsigned*)take((size_t)BB*KK*256*4);
  _Float16* fgh     = (_Float16*)take((size_t)BB*KK*256*2);
  unsigned* out_enc = (unsigned*)take((size_t)BB*KK*OUTC*4);
  _Float16* feat_h  = (_Float16*)take((size_t)BB*NN*256*2);   // 128 MB

  k_prep  <<<1024, 256, 0, stream>>>(W1, b1, g1, be1, m1, v1, W2, W3, b3, g2, be2, m2, v2, W4,
                                     W1f, b1f, b3f, W2h, W3h, W4h);
  k_init  <<<3072, 256, 0, stream>>>(fg_enc, out_enc);
  k_stage1<<<2048, 256, 0, stream>>>(xyz, choice, W1f, b1f, W2h, b2, feat_h, fg_enc);
  k_decfg <<<2048, 256, 0, stream>>>(fg_enc, fgh);
  k_stage2<<<4096, 128, 0, stream>>>(choice, feat_h, fgh, W3h, b3f, W4h, b4, out_enc);
  k_decout<<<3072, 256, 0, stream>>>(out_enc, (float*)d_out);
}
